// SHNMN_9955734192887
// MI455X (gfx1250) — matrix operand order repaired and checked by reading, not compiled, not run on hardware
//
#include <hip/hip_runtime.h>
#include <hip/hip_bf16.h>

// ---------------- problem constants ----------------
#define BN    128
#define DM    64
#define NQ    8
#define NM    3
#define E1C   36928          // D + D*D*9
#define E2C   8256           // D + 2*D*D
#define EC    (E1C + E2C)    // 45184
#define CWC   36864          // D*D*9
#define PROJC 128
#define FCD   1024
#define KSPLIT 16            // fc1 K-split factor
#define HSZ   (BN * DM * 1024)        // one (B,64,32,32) map, elements
#define XSZ   (BN * PROJC * 1024)     // one (B,128,32,32) map, elements

typedef __attribute__((ext_vector_type(16))) __bf16 v16bf;
typedef __attribute__((ext_vector_type(8)))  float  v8f;

// =====================================================================
// K0: softmax coefficients (alpha rows of 8; tau0/tau1 rows of i+2)
// coef layout: [0..23]=a_sm, [24+i*8+t]=t0_sm, [48+i*8+t]=t1_sm
// =====================================================================
__global__ void k_coeffs(const float* __restrict__ alpha,
                         const float* __restrict__ tau0,
                         const float* __restrict__ tau1,
                         float* __restrict__ coef) {
  if (threadIdx.x != 0 || blockIdx.x != 0) return;
  for (int i = 0; i < NM; ++i) {
    float mx = -1e30f;
    for (int t = 0; t < NQ; ++t) mx = fmaxf(mx, alpha[i*NQ + t]);
    float e[NQ], s = 0.f;
    for (int t = 0; t < NQ; ++t) { e[t] = __expf(alpha[i*NQ + t] - mx); s += e[t]; }
    for (int t = 0; t < NQ; ++t) coef[i*8 + t] = e[t] / s;
    int n = i + 2;
    float m0 = -1e30f, m1 = -1e30f;
    for (int t = 0; t < n; ++t) { m0 = fmaxf(m0, tau0[i*(NM+1)+t]); m1 = fmaxf(m1, tau1[i*(NM+1)+t]); }
    float e0[5], e1v[5], s0 = 0.f, s1 = 0.f;
    for (int t = 0; t < n; ++t) {
      e0[t]  = __expf(tau0[i*(NM+1)+t] - m0); s0 += e0[t];
      e1v[t] = __expf(tau1[i*(NM+1)+t] - m1); s1 += e1v[t];
    }
    for (int t = 0; t < 8; ++t) {
      coef[24 + i*8 + t] = (t < n) ? e0[t]  / s0 : 0.f;
      coef[48 + i*8 + t] = (t < n) ? e1v[t] / s1 : 0.f;
    }
  }
}

__global__ void k_zero(float* __restrict__ p, int n) {
  int i = blockIdx.x * blockDim.x + threadIdx.x;
  if (i < n) p[i] = 0.f;
}

// =====================================================================
// K1: stem conv1 (3->64, 3x3, pad1, 64x64) + relu + maxpool2 -> (B,64,32,32)
// =====================================================================
__global__ void k_stem1(const float* __restrict__ img, const float* __restrict__ w1,
                        const float* __restrict__ b1, float* __restrict__ x1) {
  int idx = blockIdx.x * blockDim.x + threadIdx.x;   // b,o,py,px
  int px = idx & 31, py = (idx >> 5) & 31, o = (idx >> 10) & 63, b = idx >> 16;
  if (b >= BN) return;
  float bias = __ldg(&b1[o]);
  float best = -1e30f;
  for (int dy = 0; dy < 2; ++dy)
    for (int dx = 0; dx < 2; ++dx) {
      int y = 2*py + dy, x = 2*px + dx;
      float s = bias;
      for (int ci = 0; ci < 3; ++ci)
        for (int ky = 0; ky < 3; ++ky) {
          int yy = y + ky - 1; if (yy < 0 || yy >= 64) continue;
          for (int kx = 0; kx < 3; ++kx) {
            int xx = x + kx - 1; if (xx < 0 || xx >= 64) continue;
            s += __ldg(&w1[(o*3 + ci)*9 + ky*3 + kx]) *
                 __ldg(&img[((b*3 + ci)*64 + yy)*64 + xx]);
          }
        }
      best = fmaxf(best, fmaxf(s, 0.f));
    }
  x1[idx] = best;
}

// =====================================================================
// K2: qrep[b,e] = sum_t a[t] * concat(emb1,emb2)[question[b,t], e]
// =====================================================================
__global__ void k_qrep(const int* __restrict__ question, const float* __restrict__ emb1,
                       const float* __restrict__ emb2, const float* __restrict__ coef,
                       int module, float* __restrict__ qrep) {
  int idx = blockIdx.x * blockDim.x + threadIdx.x;
  int b = idx / EC;
  if (b >= BN) return;
  int e = idx - b * EC;
  const float* a = coef + module * 8;
  float s = 0.f;
  #pragma unroll
  for (int t = 0; t < NQ; ++t) {
    int w = __ldg(&question[b*NQ + t]);
    float v = (e < E1C) ? __ldg(&emb1[(size_t)w * E1C + e])
                        : __ldg(&emb2[(size_t)w * E2C + (e - E1C)]);
    s += a[t] * v;
  }
  qrep[idx] = s;
}

// =====================================================================
// K3: xcat[b,c,p]: c<64 -> sum_t t0[t]*h[t][b,c,p]; c>=64 -> t1 weights
// =====================================================================
__global__ void k_xcat(const float* __restrict__ hbase, int module,
                       const float* __restrict__ coef, float* __restrict__ xcat) {
  int idx = blockIdx.x * blockDim.x + threadIdx.x;   // b,c(128),p(1024)
  int p = idx & 1023, c = (idx >> 10) & 127, b = idx >> 17;
  if (b >= BN) return;
  const float* tc = coef + ((c < 64) ? 24 : 48) + module * 8;
  int cc = c & 63;
  int nt = module + 2;
  float s = 0.f;
  for (int t = 0; t < nt; ++t)
    s += tc[t] * hbase[(size_t)t * HSZ + ((size_t)b * 64 + cc) * 1024 + p];
  xcat[idx] = s;
}

// =====================================================================
// Generic WMMA GEMM: out[b,m,p] = [relu](sum_k A[b][m,k]*Bsrc[b][k,p] + bias[b][m])
// A f32 row-major MxK (per-sample stride sA), Bsrc f32 row-major Kx1024,
// K in {64,128}. Block=128 thr (4 waves = 4 M-tiles), grid=(64, M/64, B).
// B tile staged in LDS as bf16 (converted once at staging).
// =====================================================================
__global__ __launch_bounds__(128)
void k_gemm(const float* __restrict__ Abase, long long sA,
            const float* __restrict__ biasBase, long long sBias,
            const float* __restrict__ Bbase, long long sB,
            float* __restrict__ Obase, long long sO,
            int Kd, int relu) {
  __shared__ __bf16 Bt[128 * 17];
  int b    = blockIdx.z;
  int n0   = blockIdx.x * 16;
  int wave = threadIdx.x >> 5, lane = threadIdx.x & 31;
  int m0   = blockIdx.y * 64 + wave * 16;

  const float* Bp = Bbase + (size_t)b * sB;
  for (int i = threadIdx.x; i < Kd * 16; i += blockDim.x) {
    int k = i >> 4, n = i & 15;
    Bt[k * 17 + n] = (__bf16)Bp[(size_t)k * 1024 + n0 + n];
  }
  __syncthreads();

  const float* Ap = Abase + (size_t)b * sA;
  int mrow = m0 + (lane & 15);
  int h = lane >> 4, n = lane & 15;
  v8f acc = {0.f,0.f,0.f,0.f,0.f,0.f,0.f,0.f};
  for (int k0 = 0; k0 < Kd; k0 += 32) {
    const float* arow = Ap + (size_t)mrow * Kd + k0;
    v16bf a, bb;
    #pragma unroll
    for (int j = 0; j < 8; ++j) {
      int kb = 16*h + 2*j;
      a[2*j]   = (__bf16)__ldg(arow + kb);
      a[2*j+1] = (__bf16)__ldg(arow + kb + 1);
      int kc = 16*h + 2*j;
      bb[2*j]   = Bt[(k0 + kc)     * 17 + n];
      bb[2*j+1] = Bt[(k0 + kc + 1) * 17 + n];
    }
    acc = __builtin_amdgcn_wmma_f32_16x16x32_bf16(false, a, false, bb,
                                                  (short)0, acc, false, false);
  }
  const float* bp = biasBase + (size_t)b * sBias;
  float* Op = Obase + (size_t)b * sO;
  #pragma unroll
  for (int j = 0; j < 8; ++j) {
    int m = m0 + 8*h + j;
    float v = acc[j] + __ldg(&bp[m]);
    if (relu) v = fmaxf(v, 0.f);
    Op[(size_t)m * 1024 + n0 + n] = v;
  }
}

// =====================================================================
// Conv3x3 (64->64, pad1, 32x32) as implicit WMMA GEMM.
// W row-major [o*576 + c*9 + k9] (per-sample stride sW; 0 = shared).
// Block=128 thr (4 waves = all 64 out channels), grid=(64 pixel tiles, 1, B).
// LDS: full 64x576 weight slice as bf16 (transposed [c9][o], pad 65) staged
//      once per block with coalesced global reads; input halo tile as bf16.
// =====================================================================
__global__ __launch_bounds__(128)
void k_conv3(const float* __restrict__ Wbase, long long sW,
             const float* __restrict__ biasBase, long long sBias,
             const float* __restrict__ Ibase, long long sI,
             float* __restrict__ Obase, long long sO, int relu) {
  __shared__ __bf16 Wl[576 * 65];      // [ (c*9+k9)*65 + o ]
  __shared__ __bf16 It[64 * 57];       // [c*57 + r*19 + col]
  int b    = blockIdx.z;
  int tile = blockIdx.x;               // 0..63
  int ty   = tile >> 1;
  int tx   = (tile & 1) * 16;
  int wave = threadIdx.x >> 5, lane = threadIdx.x & 31;

  const float* Wp = Wbase + (size_t)b * sW;
  for (int i = threadIdx.x; i < 64 * 576; i += blockDim.x) {
    int o = i / 576, c9 = i - o * 576;         // global reads coalesced in c9
    Wl[c9 * 65 + o] = (__bf16)__ldg(&Wp[i]);
  }
  const float* Ip = Ibase + (size_t)b * sI;
  for (int i = threadIdx.x; i < 64 * 54; i += blockDim.x) {
    int c = i / 54, rr = i - c * 54;
    int r = rr / 18, col = rr - r * 18;
    int yy = ty + r - 1, xx = tx + col - 1;
    float v = 0.f;
    if (yy >= 0 && yy < 32 && xx >= 0 && xx < 32)
      v = Ip[(size_t)c * 1024 + yy * 32 + xx];
    It[c * 57 + r * 19 + col] = (__bf16)v;
  }
  __syncthreads();

  int m0 = wave * 16;
  int mrow = m0 + (lane & 15);
  int h = lane >> 4, n = lane & 15;
  v8f acc = {0.f,0.f,0.f,0.f,0.f,0.f,0.f,0.f};
  #pragma unroll
  for (int k9 = 0; k9 < 9; ++k9) {
    int dy = k9 / 3, dx = k9 - 3 * (k9 / 3);
    #pragma unroll
    for (int ks = 0; ks < 2; ++ks) {
      int c0 = ks * 32;
      v16bf a, bb;
      #pragma unroll
      for (int j = 0; j < 8; ++j) {
        int kb = 16*h + 2*j;
        int ca = c0 + kb;
        a[2*j]   = Wl[(ca * 9 + k9)       * 65 + mrow];
        a[2*j+1] = Wl[((ca + 1) * 9 + k9) * 65 + mrow];
        int kc = c0 + 16*h + 2*j;
        bb[2*j]   = It[kc       * 57 + dy * 19 + n + dx];
        bb[2*j+1] = It[(kc + 1) * 57 + dy * 19 + n + dx];
      }
      acc = __builtin_amdgcn_wmma_f32_16x16x32_bf16(false, a, false, bb,
                                                    (short)0, acc, false, false);
    }
  }
  const float* bp = biasBase + (size_t)b * sBias;
  float* Op = Obase + (size_t)b * sO;
  #pragma unroll
  for (int j = 0; j < 8; ++j) {
    int m = m0 + 8*h + j;
    float v = acc[j] + __ldg(&bp[m]);
    if (relu) v = fmaxf(v, 0.f);
    Op[(size_t)m * 1024 + ty * 32 + tx + n] = v;
  }
}

// =====================================================================
// maxpool2 + flatten: pfull (B,128,32,32) -> flat (B, 128*16*16)
// =====================================================================
__global__ void k_pool(const float* __restrict__ pfull, float* __restrict__ flat) {
  int idx = blockIdx.x * blockDim.x + threadIdx.x;   // b, o, py, px
  int px = idx & 15, py = (idx >> 4) & 15, o = (idx >> 8) & 127, b = idx >> 15;
  if (b >= BN) return;
  const float* p = pfull + ((size_t)b * 128 + o) * 1024;
  int y = 2 * py, x = 2 * px;
  float v = fmaxf(fmaxf(p[y*32 + x], p[y*32 + x + 1]),
                  fmaxf(p[(y+1)*32 + x], p[(y+1)*32 + x + 1]));
  flat[idx] = v;
}

// =====================================================================
// fc1 partial: part[chunk,m,f] = sum_{k in chunk} flat[m,k]*fc1_w[k,f]
// grid=(64 N-tiles, 2 M-chunks, KSPLIT), block=128 (4 waves).
// Prefetch next weight slab (global_prefetch_b8 path).
// =====================================================================
__global__ __launch_bounds__(128)
void k_fc1(const float* __restrict__ flat, const float* __restrict__ w,
           float* __restrict__ part) {
  __shared__ __bf16 Bt[32 * 17];
  int n0   = blockIdx.x * 16;
  int wave = threadIdx.x >> 5, lane = threadIdx.x & 31;
  int m0   = blockIdx.y * 64 + wave * 16;
  int kbeg = blockIdx.z * (32768 / KSPLIT);
  int kend = kbeg + (32768 / KSPLIT);
  int mrow = m0 + (lane & 15);
  int h = lane >> 4, n = lane & 15;
  v8f acc = {0.f,0.f,0.f,0.f,0.f,0.f,0.f,0.f};
  for (int k0 = kbeg; k0 < kend; k0 += 32) {
    __syncthreads();
    for (int i = threadIdx.x; i < 32 * 16; i += blockDim.x) {
      int k = i >> 4, nn = i & 15;
      Bt[k * 17 + nn] = (__bf16)__ldg(&w[(size_t)(k0 + k) * 1024 + n0 + nn]);
    }
    if (k0 + 32 < kend && threadIdx.x < 32)
      __builtin_prefetch(&w[(size_t)(k0 + 32 + threadIdx.x) * 1024 + n0], 0, 3);
    __syncthreads();
    const float* arow = flat + (size_t)mrow * 32768 + k0;
    v16bf a, bb;
    #pragma unroll
    for (int j = 0; j < 8; ++j) {
      int kb = 16*h + 2*j;
      a[2*j]   = (__bf16)__ldg(arow + kb);
      a[2*j+1] = (__bf16)__ldg(arow + kb + 1);
      int kc = 16*h + 2*j;
      bb[2*j]   = Bt[kc       * 17 + n];
      bb[2*j+1] = Bt[(kc + 1) * 17 + n];
    }
    acc = __builtin_amdgcn_wmma_f32_16x16x32_bf16(false, a, false, bb,
                                                  (short)0, acc, false, false);
  }
  float* pp = part + (size_t)blockIdx.z * BN * FCD;
  #pragma unroll
  for (int j = 0; j < 8; ++j) {
    int m = m0 + 8*h + j;
    pp[(size_t)m * FCD + n0 + n] = acc[j];
  }
}

// fixed-order reduction over K-chunks + bias + relu (deterministic)
__global__ void k_fc1red(const float* __restrict__ part, const float* __restrict__ bias,
                         float* __restrict__ hfc) {
  int idx = blockIdx.x * blockDim.x + threadIdx.x;   // m*1024 + f
  if (idx >= BN * FCD) return;
  int f = idx & (FCD - 1);
  float s = __ldg(&bias[f]);
  for (int c = 0; c < KSPLIT; ++c) s += part[(size_t)c * BN * FCD + idx];
  hfc[idx] = fmaxf(s, 0.f);
}

// =====================================================================
// fc2: out[b,a] = hfc[b,:] @ fc2_w(1024x2) + b2
// =====================================================================
__global__ void k_fc2(const float* __restrict__ hfc, const float* __restrict__ w,
                      const float* __restrict__ bias, float* __restrict__ out) {
  int idx = blockIdx.x * blockDim.x + threadIdx.x;
  if (idx >= BN * 2) return;
  int a = idx & 1, b = idx >> 1;
  float s = bias[a];
  for (int k = 0; k < FCD; ++k) s += hfc[b * FCD + k] * __ldg(&w[k * 2 + a]);
  out[idx] = s;
}

// =====================================================================
static inline size_t align256(size_t x) { return (x + 255) & ~(size_t)255; }

extern "C" void kernel_launch(void* const* d_in, const int* in_sizes, int n_in,
                              void* d_out, int out_size, void* d_ws, size_t ws_size,
                              hipStream_t stream) {
  (void)in_sizes; (void)n_in; (void)out_size; (void)ws_size;
  const float* image   = (const float*)d_in[0];
  const int*   question= (const int*)  d_in[1];
  const float* emb1    = (const float*)d_in[2];
  const float* emb2    = (const float*)d_in[3];
  const float* alpha   = (const float*)d_in[4];
  const float* tau0    = (const float*)d_in[5];
  const float* tau1    = (const float*)d_in[6];
  const float* stem_w1 = (const float*)d_in[7];
  const float* stem_b1 = (const float*)d_in[8];
  const float* stem_w2 = (const float*)d_in[9];
  const float* stem_b2 = (const float*)d_in[10];
  const float* proj_w  = (const float*)d_in[11];
  const float* proj_b  = (const float*)d_in[12];
  const float* fc1_w   = (const float*)d_in[13];
  const float* fc1_b   = (const float*)d_in[14];
  const float* fc2_w   = (const float*)d_in[15];
  const float* fc2_b   = (const float*)d_in[16];
  float* out = (float*)d_out;

  char* ws = (char*)d_ws;
  size_t off = 0;
  auto take = [&](size_t elems) -> float* {
    float* p = (float*)(ws + off);
    off = align256(off + elems * sizeof(float));
    return p;
  };
  float* coef    = take(72);
  float* hbuf    = take((size_t)5 * HSZ);     // h[0..4], each (B,64,32,32)
  float* x1      = take(HSZ);                 // pooled stem1 output
  float* qrep    = take((size_t)BN * EC);     // per-module qrep; reused as `flat`
  float* proj    = take(HSZ);                 // module 1x1 proj output
  float* xcat    = take(XSZ);                 // (B,128,1024); reused as pfull
  float* hfc     = take((size_t)BN * FCD);
  float* fc1part = take((size_t)KSPLIT * BN * FCD);
  float* flat  = qrep;                        // alias: qrep dead after modules
  float* pfull = xcat;                        // alias: xcat dead after modules

  const long long sQ = EC;                    // qrep per-sample stride
  const long long sH = (long long)DM * 1024;
  const long long sX = (long long)PROJC * 1024;

  // coefficients + sentinel h0 = 0
  k_coeffs<<<1, 1, 0, stream>>>(alpha, tau0, tau1, coef);
  k_zero<<<(HSZ + 255) / 256, 256, 0, stream>>>(hbuf, HSZ);

  // stem conv1 + relu + pool -> x1
  k_stem1<<<(BN * DM * 1024) / 256, 256, 0, stream>>>(image, stem_w1, stem_b1, x1);

  // stem conv2 (shared weights) + relu -> h[1]
  k_conv3<<<dim3(64, 1, BN), 128, 0, stream>>>(
      stem_w2, 0, stem_b2, 0, x1, sH, hbuf + (size_t)1 * HSZ, sH, 1);

  // modules
  for (int i = 0; i < NM; ++i) {
    k_qrep<<<((size_t)BN * EC + 255) / 256, 256, 0, stream>>>(
        question, emb1, emb2, coef, i, qrep);
    k_xcat<<<(BN * PROJC * 1024) / 256, 256, 0, stream>>>(hbuf, i, coef, xcat);
    // per-sample 1x1 proj: A=qrep[:,36928:45120] (64x128), bias=qrep[:,45120:]
    k_gemm<<<dim3(64, 1, BN), 128, 0, stream>>>(
        qrep + (CWC + DM), sQ, qrep + (CWC + DM + 2 * DM * DM), sQ,
        xcat, sX, proj, sH, /*K=*/2 * DM, /*relu=*/0);
    // per-sample 3x3 conv: W=qrep[:, :36864], bias=qrep[:,36864:36928]
    k_conv3<<<dim3(64, 1, BN), 128, 0, stream>>>(
        qrep, sQ, qrep + CWC, sQ, proj, sH,
        hbuf + (size_t)(i + 2) * HSZ, sH, /*relu=*/0);
  }

  // final projection (shared 128x64 weights) + relu -> pfull (B,128,32,32)
  k_gemm<<<dim3(64, 2, BN), 128, 0, stream>>>(
      proj_w, 0, proj_b, 0,
      hbuf + (size_t)4 * HSZ, sH, pfull, sX, /*K=*/DM, /*relu=*/1);

  // maxpool + flatten -> flat (B,32768)
  k_pool<<<(BN * 32768) / 256, 256, 0, stream>>>(pfull, flat);

  // fc1 (WMMA, K split 16 ways) -> partials -> reduce -> hfc ; fc2 -> out
  k_fc1<<<dim3(64, 2, KSPLIT), 128, 0, stream>>>(flat, fc1_w, fc1part);
  k_fc1red<<<(BN * FCD + 255) / 256, 256, 0, stream>>>(fc1part, fc1_b, hfc);
  k_fc2<<<1, 256, 0, stream>>>(hfc, fc2_w, fc2_b, out);
}
